// TensorProduct_35347580846607
// MI455X (gfx1250) — matrix operand order repaired and checked by reading, not compiled, not run on hardware
//
#include <hip/hip_runtime.h>
#include <math.h>

// ---------------------------------------------------------------------------
// Types
// ---------------------------------------------------------------------------
typedef __attribute__((ext_vector_type(16))) __bf16        v16bf;
typedef __attribute__((ext_vector_type(8)))  float         v8f;
typedef __attribute__((ext_vector_type(4)))  unsigned int  u32x4;

union BF16Frag { v16bf v; u32x4 q[2]; };
union BF16Pack { __bf16 h[8]; u32x4 q; };

// Path tables (order matches reference PATHS enumeration):
//  0:(0,0,0) 1:(0,1,1) 2:(0,2,2) 3:(1,0,1) 4:(1,1,0) 5:(1,1,1) 6:(1,1,2)
//  7:(1,2,1) 8:(1,2,2) 9:(2,0,2) 10:(2,1,1) 11:(2,1,2) 12:(2,2,0)
//  13:(2,2,1) 14:(2,2,2)
__device__ __constant__ int c_L1[15]  = {0,0,0,1,1,1,1,1,1,2,2,2,2,2,2};
__device__ __constant__ int c_L2[15]  = {0,1,2,0,1,1,1,2,2,0,1,1,2,2,2};
__device__ __constant__ int c_L3[15]  = {0,1,2,1,0,1,2,1,2,2,1,2,0,1,2};
// dense CG offsets, sizes (2l1+1)(2l2+1)(2l3+1); total 615 floats
__device__ __constant__ int c_CGOFF[15] =
    {0,1,10,35,44,53,80,125,170,245,270,315,390,415,490};

#define CG_FLOATS   615
#define WS_CG_BYTES 4096              // CG region in d_ws
#define NFRAG       (15*4*2)          // path x wtile x ustep
#define FRAG_BYTES  1024              // 32 lanes * 16 bf16 * 2B

// Z buffer: up to 18 columns x 16 samples x 72 bf16 (8 pad) = 41472 bytes.
#define ZROW_BF16   72
#define ZBUF_BYTES  (18*16*ZROW_BF16*2)
#define SMEM_BYTES  (ZBUF_BYTES + ((CG_FLOATS*4 + 15) & ~15))

// ---------------------------------------------------------------------------
// Kernel 1: compute real-basis CG tensors (matches reference _real_cg)
// ---------------------------------------------------------------------------
struct cdbl { double re, im; };
__device__ __forceinline__ cdbl cmul(cdbl a, cdbl b) {
    return cdbl{a.re*b.re - a.im*b.im, a.re*b.im + a.im*b.re};
}
__device__ __forceinline__ double dfact(int n) {
    double r = 1.0;
    for (int i = 2; i <= n; ++i) r *= (double)i;
    return r;
}
__device__ double su2cg(int j1,int m1,int j2,int m2,int j3,int m3) {
    if (m3 != m1 + m2) return 0.0;
    double c = (2.0*j3+1.0) * dfact(j1+j2-j3)*dfact(j1-j2+j3)*dfact(-j1+j2+j3)
               / dfact(j1+j2+j3+1);
    c = sqrt(c);
    c *= sqrt(dfact(j3+m3)*dfact(j3-m3)*dfact(j1-m1)*dfact(j1+m1)
              *dfact(j2-m2)*dfact(j2+m2));
    int k0 = 0;
    if (j2-j3-m1 > k0) k0 = j2-j3-m1;
    if (j1-j3+m2 > k0) k0 = j1-j3+m2;
    int k1 = j1+j2-j3;
    if (j1-m1 < k1) k1 = j1-m1;
    if (j2+m2 < k1) k1 = j2+m2;
    double s = 0.0;
    for (int k = k0; k <= k1; ++k) {
        double d = dfact(k)*dfact(j1+j2-j3-k)*dfact(j1-m1-k)*dfact(j2+m2-k)
                   *dfact(j3-j2+m1+k)*dfact(j3-j1-m2+k);
        s += ((k & 1) ? -1.0 : 1.0) / d;
    }
    return c * s;
}
__device__ void qmat(int l, cdbl* q) {   // q[(2l+1)*(2l+1)], row = complex m
    int d = 2*l + 1;
    for (int i = 0; i < d*d; ++i) q[i] = cdbl{0.0, 0.0};
    double r2 = 1.0 / sqrt(2.0);
    for (int m = -l; m < 0; ++m) {
        q[(l+m)*d + (l-m)] = cdbl{r2, 0.0};
        q[(l+m)*d + (l+m)] = cdbl{0.0, -r2};
    }
    q[l*d + l] = cdbl{1.0, 0.0};
    for (int m = 1; m <= l; ++m) {
        double sg = (m & 1) ? -1.0 : 1.0;
        q[(l+m)*d + (l+m)] = cdbl{sg*r2, 0.0};
        q[(l+m)*d + (l-m)] = cdbl{0.0, sg*r2};
    }
    cdbl f = (l == 0) ? cdbl{1.0, 0.0} : (l == 1) ? cdbl{0.0, -1.0}
                                                  : cdbl{-1.0, 0.0};
    for (int i = 0; i < d*d; ++i) q[i] = cmul(f, q[i]);
}

__global__ void cg_init_kernel(float* __restrict__ cgout) {
    __shared__ cdbl   U1s[25], U2s[25], U3s[25];
    __shared__ double Cs[125], Crs[125], red[128];
    const int p = blockIdx.x;
    const int t = threadIdx.x;
    const int l1 = c_L1[p], l2 = c_L2[p], l3 = c_L3[p];
    const int d1 = 2*l1+1, d2 = 2*l2+1, d3 = 2*l3+1;
    const int ne = d1*d2*d3;

    if (t == 0) qmat(l1, U1s);
    else if (t == 1) qmat(l2, U2s);
    else if (t == 2) qmat(l3, U3s);
    __syncthreads();

    for (int e = t; e < ne; e += 128) {
        int i = e / (d2*d3), j = (e / d3) % d2, k = e % d3;
        Cs[e] = su2cg(l1, i-l1, l2, j-l2, l3, k-l3);
    }
    __syncthreads();

    for (int e = t; e < ne; e += 128) {
        int a = e / (d2*d3), b = (e / d3) % d2, c = e % d3;
        double acc = 0.0;
        for (int i = 0; i < d1; ++i)
            for (int j = 0; j < d2; ++j)
                for (int k = 0; k < d3; ++k) {
                    double cr = Cs[(i*d2 + j)*d3 + k];
                    if (cr == 0.0) continue;
                    cdbl tt = cmul(U1s[i*d1 + a], U2s[j*d2 + b]);
                    cdbl u3 = cdbl{U3s[k*d3 + c].re, -U3s[k*d3 + c].im};
                    tt = cmul(tt, u3);
                    acc += cr * tt.re;
                }
        Crs[e] = acc;
    }
    __syncthreads();

    double s = 0.0;
    for (int e = t; e < ne; e += 128) s += Crs[e] * Crs[e];
    red[t] = s;
    __syncthreads();
    for (int w = 64; w > 0; w >>= 1) {
        if (t < w) red[t] += red[t + w];
        __syncthreads();
    }
    double inv = 1.0 / sqrt(red[0]);
    for (int e = t; e < ne; e += 128)
        cgout[c_CGOFF[p] + e] = (float)(Crs[e] * inv);
}

// ---------------------------------------------------------------------------
// Kernel 2: pack W (alpha folded) into bf16 WMMA A-fragment order.
// Fragment F = (p*4 + wtile)*2 + ustep; element index within fragment:
// chunk*256 + lane*8 + el  (e = chunk*8+el), A[m=lane%16, k] with
// k = e + (e<8?0:8) + (lane<16?0:8)   (ISA 16-bit A 16x32 layout).
// ---------------------------------------------------------------------------
__global__ void wfrag_build_kernel(const float* __restrict__ W,
                                   __bf16* __restrict__ frag) {
    int idx = blockIdx.x * 256 + threadIdx.x;
    if (idx >= NFRAG * 512) return;
    int f      = idx >> 9;
    int within = idx & 511;
    int chunk  = within >> 8;
    int rem    = within & 255;
    int ln     = rem >> 3;
    int el     = rem & 7;
    int e      = chunk*8 + el;
    int p      = f >> 3;
    int wtile  = (f >> 1) & 3;
    int ustep  = f & 1;
    int m      = ln & 15;
    int k      = e + ((e < 8) ? 0 : 8) + ((ln < 16) ? 0 : 8);
    int u      = ustep*32 + k;
    int w      = wtile*16 + m;
    const int npt[3] = {3, 6, 6};                 // paths feeding each l3
    float alpha = rsqrtf(64.0f * (float)npt[c_L3[p]]);
    frag[idx] = (__bf16)(W[p*4096 + u*64 + w] * alpha);
}

// ---------------------------------------------------------------------------
// Kernel 3: main fused TP kernel. 128 threads = 4 waves; wave = w-tile.
// Per block: 16 samples. Z staged in LDS (bf16, B-fragment rows), WMMA
// v_wmma_f32_16x16x32_bf16 with f32 accumulation, C staged through LDS for
// coalesced f32 stores.
// ---------------------------------------------------------------------------
template<int L1, int L2, int L3>
__device__ __forceinline__ void zstage_path(const float (&xr)[8][9],
                                            const float (&yv)[9],
                                            const float* cgl, int cgoff,
                                            unsigned char* smem,
                                            int colbase, int n, int ublk) {
    constexpr int D1 = 2*L1+1, D2 = 2*L2+1, D3 = 2*L3+1;
    constexpr int XO = (L1 == 0) ? 0 : (L1 == 1 ? 1 : 4);
    constexpr int YO = (L2 == 0) ? 0 : (L2 == 1 ? 1 : 4);
    float ycg[D1][D3];
#pragma unroll
    for (int i = 0; i < D1; ++i)
#pragma unroll
        for (int k = 0; k < D3; ++k) {
            float s = 0.f;
#pragma unroll
            for (int j = 0; j < D2; ++j)
                s += yv[YO + j] * cgl[cgoff + (i*D2 + j)*D3 + k];
            ycg[i][k] = s;
        }
#pragma unroll
    for (int k = 0; k < D3; ++k) {
        BF16Pack pk;
#pragma unroll
        for (int u8 = 0; u8 < 8; ++u8) {
            float s = 0.f;
#pragma unroll
            for (int i = 0; i < D1; ++i)
                s += xr[u8][XO + i] * ycg[i][k];
            pk.h[u8] = (__bf16)s;
        }
        int col = colbase + k;
        ((u32x4*)smem)[(col*16 + n)*9 + ublk] = pk.q;   // 16B ds_store
    }
}

template<int D3, int NP>
__device__ __forceinline__ void wmma_group(const u32x4* __restrict__ wq,
                                           const u32x4* zq,
                                           const int (&pg)[NP],
                                           int wave, int lane, v8f* acc) {
    const int nl  = lane & 15;
    const int sel = lane >> 4;
#pragma unroll
    for (int pl = 0; pl < NP; ++pl) {
        const int p = pg[pl];
#pragma unroll
        for (int us = 0; us < 2; ++us) {
            const int fb = ((p*4 + wave)*2 + us) * 64;   // uint4 units
            BF16Frag a;
            a.q[0] = wq[fb + lane];
            a.q[1] = wq[fb + 32 + lane];
#pragma unroll
            for (int kk = 0; kk < D3; ++kk) {
                const int col = pl*D3 + kk;
                const int zb  = (col*16 + nl)*9 + us*4 + sel;
                BF16Frag b;
                b.q[0] = zq[zb];
                b.q[1] = zq[zb + 2];
                acc[kk] = __builtin_amdgcn_wmma_f32_16x16x32_bf16(
                    false, a.v, false, b.v, (short)0, acc[kk], false, false);
            }
        }
    }
}

template<int D3>
__device__ __forceinline__ void dump_out(unsigned char* smem,
                                         float* __restrict__ out,
                                         const v8f* acc, int off3,
                                         int wave, int lane, int g, int tid) {
    float* cb = (float*)smem;                  // C staging aliases Z buffer
    const int nl  = lane & 15;
    const int sel = lane >> 4;
#pragma unroll
    for (int kk = 0; kk < D3; ++kk)
#pragma unroll
        for (int r = 0; r < 8; ++r)
            cb[((kk*64 + wave*16 + sel*8 + r) << 4) + nl] = acc[kk][r];
    __syncthreads();
    constexpr int SEG = 64 * D3;
    for (int idx = tid; idx < 16*SEG; idx += 128) {
        int n = idx / SEG, pos = idx - n*SEG;
        int w = pos / D3,  k   = pos - w*D3;
        out[(g*16 + n)*576 + off3 + pos] = cb[((k*64 + w) << 4) + n];
    }
    __syncthreads();
}

__global__ __launch_bounds__(128)
void tp_main_kernel(const float* __restrict__ x, const float* __restrict__ y,
                    const float* __restrict__ cg,
                    const u32x4* __restrict__ wq, float* __restrict__ out) {
    __shared__ __align__(16) unsigned char smem[SMEM_BYTES];
    const int tid  = threadIdx.x;
    const int g    = blockIdx.x;
    const int n    = tid & 15;
    const int ublk = tid >> 4;       // 0..7 : owns u = ublk*8 .. +7
    const int lane = tid & 31;
    const int wave = tid >> 5;       // w-tile

    // CG -> LDS (broadcast reads later)
    float* cgl = (float*)(smem + ZBUF_BYTES);
    for (int i = tid; i < CG_FLOATS; i += 128) cgl[i] = cg[i];

    // per-thread x registers: [u8][ 0 | 1..3 | 4..8 ] = l=0,1,2 blocks
    const int ng = g*16 + n;
    const float* xrow = x + ng*576;
    float xr[8][9];
#pragma unroll
    for (int u8 = 0; u8 < 8; ++u8) {
        int u = ublk*8 + u8;
        xr[u8][0] = xrow[u];
#pragma unroll
        for (int i = 0; i < 3; ++i) xr[u8][1 + i] = xrow[64  + u*3 + i];
#pragma unroll
        for (int i = 0; i < 5; ++i) xr[u8][4 + i] = xrow[256 + u*5 + i];
    }
    float yv[9];
#pragma unroll
    for (int i = 0; i < 9; ++i) yv[i] = y[ng*9 + i];
    __syncthreads();

    const u32x4* zq = (const u32x4*)smem;
    v8f acc[5];
    const v8f vz = {0.f,0.f,0.f,0.f,0.f,0.f,0.f,0.f};

    // ------------------ l3 = 0 (paths 0,4,12), 3 cols ------------------
    zstage_path<0,0,0>(xr, yv, cgl, c_CGOFF[0],  smem, 0, n, ublk);
    zstage_path<1,1,0>(xr, yv, cgl, c_CGOFF[4],  smem, 1, n, ublk);
    zstage_path<2,2,0>(xr, yv, cgl, c_CGOFF[12], smem, 2, n, ublk);
    __syncthreads();
    acc[0] = vz;
    {
        constexpr int pg[3] = {0, 4, 12};
        wmma_group<1,3>(wq, zq, pg, wave, lane, acc);
    }
    __syncthreads();
    dump_out<1>(smem, out, acc, 0, wave, lane, g, tid);

    // ------------------ l3 = 1 (6 paths), 18 cols ----------------------
    zstage_path<0,1,1>(xr, yv, cgl, c_CGOFF[1],  smem,  0, n, ublk);
    zstage_path<1,0,1>(xr, yv, cgl, c_CGOFF[3],  smem,  3, n, ublk);
    zstage_path<1,1,1>(xr, yv, cgl, c_CGOFF[5],  smem,  6, n, ublk);
    zstage_path<1,2,1>(xr, yv, cgl, c_CGOFF[7],  smem,  9, n, ublk);
    zstage_path<2,1,1>(xr, yv, cgl, c_CGOFF[10], smem, 12, n, ublk);
    zstage_path<2,2,1>(xr, yv, cgl, c_CGOFF[13], smem, 15, n, ublk);
    __syncthreads();
    acc[0] = vz; acc[1] = vz; acc[2] = vz;
    {
        constexpr int pg[6] = {1, 3, 5, 7, 10, 13};
        wmma_group<3,6>(wq, zq, pg, wave, lane, acc);
    }
    __syncthreads();
    dump_out<3>(smem, out, acc, 64, wave, lane, g, tid);

    // ------------------ l3 = 2 (6 paths, split 3+3), 15 cols each ------
    zstage_path<0,2,2>(xr, yv, cgl, c_CGOFF[2],  smem,  0, n, ublk);
    zstage_path<1,1,2>(xr, yv, cgl, c_CGOFF[6],  smem,  5, n, ublk);
    zstage_path<1,2,2>(xr, yv, cgl, c_CGOFF[8],  smem, 10, n, ublk);
    __syncthreads();
    acc[0] = vz; acc[1] = vz; acc[2] = vz; acc[3] = vz; acc[4] = vz;
    {
        constexpr int pg[3] = {2, 6, 8};
        wmma_group<5,3>(wq, zq, pg, wave, lane, acc);
    }
    __syncthreads();
    zstage_path<2,0,2>(xr, yv, cgl, c_CGOFF[9],  smem,  0, n, ublk);
    zstage_path<2,1,2>(xr, yv, cgl, c_CGOFF[11], smem,  5, n, ublk);
    zstage_path<2,2,2>(xr, yv, cgl, c_CGOFF[14], smem, 10, n, ublk);
    __syncthreads();
    {
        constexpr int pg[3] = {9, 11, 14};
        wmma_group<5,3>(wq, zq, pg, wave, lane, acc);   // keeps accumulating
    }
    __syncthreads();
    dump_out<5>(smem, out, acc, 256, wave, lane, g, tid);
}

// ---------------------------------------------------------------------------
// Launcher
// ---------------------------------------------------------------------------
extern "C" void kernel_launch(void* const* d_in, const int* in_sizes, int n_in,
                              void* d_out, int out_size, void* d_ws,
                              size_t ws_size, hipStream_t stream) {
    (void)n_in; (void)out_size; (void)ws_size;
    const float* x = (const float*)d_in[0];
    const float* y = (const float*)d_in[1];
    const float* W = (const float*)d_in[2];
    float*  out  = (float*)d_out;
    float*  cg   = (float*)d_ws;
    __bf16* frag = (__bf16*)((char*)d_ws + WS_CG_BYTES);

    const int nb = in_sizes[0] / 576;     // 32768 samples

    cg_init_kernel<<<15, 128, 0, stream>>>(cg);
    wfrag_build_kernel<<<(NFRAG*512 + 255)/256, 256, 0, stream>>>(W, frag);
    tp_main_kernel<<<nb/16, 128, 0, stream>>>(x, y, cg,
                                              (const u32x4*)frag, out);
}
